// DeepWalk_HierSoftmax_14568529068250
// MI455X (gfx1250) — matrix operand order repaired and checked by reading, not compiled, not run on hardware
//
#include <hip/hip_runtime.h>
#include <hip/hip_bf16.h>

// ---------------------------------------------------------------------------
// DeepWalk hierarchical-softmax loss, MI455X (gfx1250, wave32, WMMA).
//
// Bandwidth-bound (~640 MB gathered / 23.3 TB/s ~= 27 us floor). One wave32
// per batch element. Dots computed with v_wmma_f32_16x16x32_f16: 16 tree
// levels per 16x16 tile (B = center embedding broadcast across the 16
// columns), f32->f16 convert on the fly, f32 accumulation inside the WMMA.
// ---------------------------------------------------------------------------

typedef __attribute__((ext_vector_type(16))) _Float16 v16h;
typedef __attribute__((ext_vector_type(8)))  float    v8f;

#define NUM_NODES (1u << 19)   // 524288 (power of two -> path length 19)
#define DIM       128
#define WAVES_PER_BLOCK 8

// A-operand chunk (16-bit A 16x32 layout, ISA 7.12.2):
//   lanes 0-15  (koff=0): VGPR0-3 = K 0..7,  VGPR4-7 = K 16..23   (of 32-chunk)
//   lanes 16-31 (koff=8): VGPR0-3 = K 8..15, VGPR4-7 = K 24..31
// v16h element j lives in VGPR j/2 (half j%2), so elements 0..7 are the low
// 8-K segment and 8..15 the +16 segment.
__device__ __forceinline__ v16h pack_a_chunk(const float* __restrict__ row,
                                             int kbase, int koff) {
  const float4* p0 = reinterpret_cast<const float4*>(row + kbase + koff);
  const float4* p1 = reinterpret_cast<const float4*>(row + kbase + 16 + koff);
  float4 q0 = p0[0], q1 = p0[1], q2 = p1[0], q3 = p1[1];
  v16h h;
  h[0]  = (_Float16)q0.x; h[1]  = (_Float16)q0.y; h[2]  = (_Float16)q0.z; h[3]  = (_Float16)q0.w;
  h[4]  = (_Float16)q1.x; h[5]  = (_Float16)q1.y; h[6]  = (_Float16)q1.z; h[7]  = (_Float16)q1.w;
  h[8]  = (_Float16)q2.x; h[9]  = (_Float16)q2.y; h[10] = (_Float16)q2.z; h[11] = (_Float16)q2.w;
  h[12] = (_Float16)q3.x; h[13] = (_Float16)q3.y; h[14] = (_Float16)q3.z; h[15] = (_Float16)q3.w;
  return h;
}

// B-operand chunk (contiguous K order, unused; B is built with pack_a_chunk):
//   lanes 0-15 hold K = kbase+0..15, lanes 16-31 hold K = kbase+16..31,
//   element j of v16h <-> K = base + j. All 16 columns identical (= e chunk).
__device__ __forceinline__ v16h pack_b_chunk(const float* e, int base) {
  const float4* p = reinterpret_cast<const float4*>(e + base);
  float4 q0 = p[0], q1 = p[1], q2 = p[2], q3 = p[3];
  v16h h;
  h[0]  = (_Float16)q0.x; h[1]  = (_Float16)q0.y; h[2]  = (_Float16)q0.z; h[3]  = (_Float16)q0.w;
  h[4]  = (_Float16)q1.x; h[5]  = (_Float16)q1.y; h[6]  = (_Float16)q1.z; h[7]  = (_Float16)q1.w;
  h[8]  = (_Float16)q2.x; h[9]  = (_Float16)q2.y; h[10] = (_Float16)q2.z; h[11] = (_Float16)q2.w;
  h[12] = (_Float16)q3.x; h[13] = (_Float16)q3.y; h[14] = (_Float16)q3.z; h[15] = (_Float16)q3.w;
  return h;
}

// Numerically stable log(sigmoid(x)) = min(x,0) - log1p(exp(-|x|))
__device__ __forceinline__ float log_sigmoid(float x) {
  return fminf(x, 0.0f) - log1pf(__expf(-fabsf(x)));
}

__global__ __launch_bounds__(WAVES_PER_BLOCK * 32)
void hier_softmax_kernel(const int* __restrict__ center,
                         const int* __restrict__ context,
                         const float* __restrict__ emb,
                         const float* __restrict__ probs,
                         float* __restrict__ out, int batch) {
  __shared__ float elds[WAVES_PER_BLOCK * DIM];

  const int lane = threadIdx.x & 31;
  const int wave = threadIdx.x >> 5;
  int b = blockIdx.x * WAVES_PER_BLOCK + wave;
  if (b >= batch) b = batch - 1;   // wave-uniform clamp; keeps EXEC all-ones

  const int      c0   = center[b];
  const unsigned cn   = (unsigned)context[b] + NUM_NODES;   // leaf id, < 2^20
  const unsigned long long cn64 = cn;

  // Stage center embedding through LDS once (avoids 16x redundant global
  // reads of the 512B row when building the broadcast B operand).
  const float* erow = emb + (size_t)c0 * DIM;
  float* ew = &elds[wave * DIM];
  reinterpret_cast<float4*>(ew)[lane] =
      reinterpret_cast<const float4*>(erow)[lane];
  __syncthreads();

  const int m    = lane & 15;       // tile row this lane feeds
  const int hi   = lane >> 4;       // 0: lanes 0-15, 1: lanes 16-31
  const int koff = hi * 8;          // A-operand K sub-offset

  // Tile 1 rows = tree levels 1..16; tile 2 rows 0,1 = levels 17,18
  // (tile-2 rows >= 2 are unused garbage; shifted ids stay < 2^20 so the
  //  gathers remain in-bounds in probs_tensor[2^20][128]).
  const unsigned node1 = (unsigned)(cn64 >> (m + 1));
  const unsigned node2 = (unsigned)(cn64 >> (m + 17));
  const float* rowA1 = probs + (size_t)node1 * DIM;
  const float* rowA2 = probs + (size_t)node2 * DIM;

  v8f acc1 = {};   // levels 1..16 dot accumulators
  v8f acc2 = {};   // levels 17..18 (rows 0,1)
#pragma unroll
  for (int c = 0; c < 4; ++c) {               // K = 128 in 4 chunks of 32
    const int kbase = c * 32;
    v16h bfrag = pack_a_chunk(ew, kbase, koff);
    v16h a1 = pack_a_chunk(rowA1, kbase, koff);
    acc1 = __builtin_amdgcn_wmma_f32_16x16x32_f16(
        false, a1, false, bfrag, (short)0, acc1, false, false);
    v16h a2 = pack_a_chunk(rowA2, kbase, koff);
    acc2 = __builtin_amdgcn_wmma_f32_16x16x32_f16(
        false, a2, false, bfrag, (short)0, acc2, false, false);
  }

  // C/D layout: lanes 0-15 VGPR r -> M=r (levels 1+r); lanes 16-31 -> M=8+r
  // (levels 9+r). All 16 columns are identical, so every lane of a half
  // holds the same dots. sign_i = +1 iff bit i of the leaf id is 0.
  float s = 0.0f;
  const int ibase = 1 + hi * 8;
#pragma unroll
  for (int r = 0; r < 8; ++r) {
    const int   i    = ibase + r;
    const float sign = ((cn >> i) & 1u) ? -1.0f : 1.0f;
    s += log_sigmoid(sign * acc1[r]);
  }
#pragma unroll
  for (int r = 0; r < 2; ++r) {               // levels 17,18 live in low half
    const int   i    = 17 + r;
    const float sign = ((cn >> i) & 1u) ? -1.0f : 1.0f;
    const float v    = log_sigmoid(sign * acc2[r]);
    s += (hi == 0) ? v : 0.0f;                // v_cndmask, no EXEC change
  }

  // Combine low half (levels 1-8,17,18) with high half (levels 9-16).
  const float total = s + __shfl_xor(s, 16, 32);
  if (lane == 0) out[b] = -total;
}

extern "C" void kernel_launch(void* const* d_in, const int* in_sizes, int n_in,
                              void* d_out, int out_size, void* d_ws, size_t ws_size,
                              hipStream_t stream) {
  const int*   center  = (const int*)d_in[0];
  const int*   context = (const int*)d_in[1];
  const float* emb     = (const float*)d_in[2];
  const float* probs   = (const float*)d_in[3];
  float*       out     = (float*)d_out;

  const int batch  = in_sizes[0];                       // 65536
  const int blocks = (batch + WAVES_PER_BLOCK - 1) / WAVES_PER_BLOCK;
  hier_softmax_kernel<<<blocks, WAVES_PER_BLOCK * 32, 0, stream>>>(
      center, context, emb, probs, out, batch);
}
